// Attention_45148696216940
// MI455X (gfx1250) — matrix operand order repaired and checked by reading, not compiled, not run on hardware
//
#include <hip/hip_runtime.h>
#include <hip/hip_bf16.h>

// ---------------------------------------------------------------------------
// Spatial self-attention, B=4 C=256 N=4096, softmax over axis=1 (columns).
// All GEMMs on v_wmma_f32_16x16x32_bf16 (f32 accumulate). Never materializes
// the [B,N,N] attention matrix: column softmax stats (m[j], D[j]) are computed
// in a streaming pass, then S is recomputed tile-by-tile for the P*V GEMM.
// ---------------------------------------------------------------------------

typedef __attribute__((ext_vector_type(16))) __bf16 bf16x16;
typedef __attribute__((ext_vector_type(8)))  float  f32x8;

#define CCH   256
#define NPIX  4096
#define NBATCH 4
#define SCALE 0.0625f   // 1/sqrt(256)

union FragU { bf16x16 v; unsigned int u[8]; uint4 q[2]; };

__device__ inline f32x8 wmma_bf16(const bf16x16& a, const bf16x16& b, const f32x8& c) {
  return __builtin_amdgcn_wmma_f32_16x16x32_bf16(false, a, false, b, (short)0, c,
                                                 false, false);
}

// A-operand fragment (16x32 bf16, M = lane&15), source row-major [M][K].
// ISA layout: VGPR p holds K = (p>>2)*16 + (lane>>4)*8 + (p&3)*2 (+1).
__device__ inline bf16x16 load_fragA(const __bf16* base, int ld) {
  int lane = threadIdx.x & 31;
  int r = lane & 15, hi = lane >> 4;
  const __bf16* row = base + (size_t)r * ld + hi * 8;
  FragU f;
#pragma unroll
  for (int g = 0; g < 2; ++g)
#pragma unroll
    for (int t = 0; t < 4; ++t)
      f.u[g * 4 + t] = *reinterpret_cast<const unsigned int*>(row + g * 16 + t * 2);
  return f.v;
}

// B-operand fragment (32x16 bf16, N = lane&15), source row-major [N][K].
// ISA layout: VGPR p holds K = (p>>2)*16 + (lane>>4)*8 + (p&3)*2 (+1).
__device__ inline bf16x16 load_fragB(const __bf16* base, int ld) {
  int lane = threadIdx.x & 31;
  int n = lane & 15, hi = lane >> 4;
  const __bf16* row = base + (size_t)n * ld + hi * 8;
  FragU f;
#pragma unroll
  for (int t = 0; t < 8; ++t)
    f.u[t] = *reinterpret_cast<const unsigned int*>(row + (t >> 2) * 16 + (t & 3) * 2);
  return f.v;
}

// Pre-packed weight fragment: lane's 16 bf16 contiguous -> two b128 loads.
__device__ inline bf16x16 load_fragPacked(const __bf16* base) {
  int lane = threadIdx.x & 31;
  const uint4* p = reinterpret_cast<const uint4*>(base + (size_t)lane * 16);
  FragU f;
  f.q[0] = p[0];
  f.q[1] = p[1];
  return f.v;
}

// ---------------------------------------------------------------------------
// Kernel 0: pack one 256x256 f32 weight matrix into A-fragment-ordered bf16.
// grid = 128 tiles (mo 0..15 x kb 0..7), block = 32 lanes.
// ---------------------------------------------------------------------------
__global__ __launch_bounds__(32) void pack_w_kernel(const float* __restrict__ w,
                                                    __bf16* __restrict__ out) {
  int tile = blockIdx.x;              // mo*8 + kb
  int mo = tile >> 3, kb = tile & 7;
  int lane = threadIdx.x;
  int r = lane & 15, hi = lane >> 4;
  const float* src = w + (size_t)(mo * 16 + r) * CCH + kb * 32 + hi * 8;
  __bf16* dst = out + ((size_t)tile * 32 + lane) * 16;
#pragma unroll
  for (int g = 0; g < 2; ++g)
#pragma unroll
    for (int t = 0; t < 8; ++t)
      dst[g * 8 + t] = (__bf16)src[g * 16 + t];
}

// ---------------------------------------------------------------------------
// One 32x... projection GEMM block for a single weight matrix.
// MODE 0/1: store transposed [B][N][C] (contiguous over o -> b128 merge).
// MODE 2:   store [B][C][N] (v layout for the later P*V B-operand).
// Static pointers => global_load/global_store; template => branch-free stores.
// ---------------------------------------------------------------------------
template <int MODE>
__device__ __forceinline__ void proj_one_matrix(
    const __bf16* __restrict__ wm, const float* __restrict__ bias,
    __bf16* __restrict__ dst, const bf16x16 (&bfr)[8],
    int b, int n, int wave, int hi, int ln) {
#pragma unroll 1
  for (int oo = 0; oo < 2; ++oo) {
    int ot = wave * 2 + oo;
    f32x8 acc = {0.f, 0.f, 0.f, 0.f, 0.f, 0.f, 0.f, 0.f};
#pragma unroll
    for (int kb = 0; kb < 8; ++kb) {
      bf16x16 af = load_fragPacked(wm + (size_t)(ot * 8 + kb) * 512);
      acc = wmma_bf16(af, bfr[kb], acc);
    }
#pragma unroll
    for (int r = 0; r < 8; ++r) {
      int o = ot * 16 + hi * 8 + r;
      float val = acc[r] + bias[o];
      if (MODE == 2)
        dst[((size_t)b * CCH + o) * NPIX + n] = (__bf16)val;
      else
        dst[((size_t)b * NPIX + n) * CCH + o] = (__bf16)val;
    }
  }
}

// ---------------------------------------------------------------------------
// Kernel 1: QKV projection. grid = BN/64 = 256 blocks of 256 threads (8 waves).
// Stores qT,kT as [B][N][C] bf16 (for A/B fragment loads) and v as [B][C][N].
// sched_barrier between the three matrices keeps only one set of A-fragments
// live at a time (no spills), while each block keeps static store paths.
// ---------------------------------------------------------------------------
__global__ __launch_bounds__(256) void qkv_kernel(
    const float* __restrict__ x,
    const __bf16* __restrict__ wpq, const __bf16* __restrict__ wpk,
    const __bf16* __restrict__ wpv,
    const float* __restrict__ bq, const float* __restrict__ bk,
    const float* __restrict__ bv,
    __bf16* __restrict__ qT, __bf16* __restrict__ kT, __bf16* __restrict__ vB) {
  __shared__ __bf16 xt[64 * 264];               // x tile transposed [n][c], padded
  int tid = threadIdx.x;
  int pix0 = blockIdx.x * 64;
  int b = pix0 >> 12;
  int n0 = pix0 & (NPIX - 1);
  const float* xb = x + (size_t)b * CCH * NPIX;
#pragma unroll 4
  for (int rep = 0; rep < 64; ++rep) {          // coalesced load + LDS transpose
    int idx = rep * 256 + tid;
    int c = idx >> 6, n = idx & 63;
    xt[n * 264 + c] = (__bf16)xb[(size_t)c * NPIX + n0 + n];
  }
  __syncthreads();

  int lane = tid & 31, wave = tid >> 5;
  int hi = lane >> 4, ln = lane & 15;

#pragma unroll 1
  for (int nt = 0; nt < 4; ++nt) {
    bf16x16 bfr[8];
#pragma unroll
    for (int kb = 0; kb < 8; ++kb)
      bfr[kb] = load_fragB(xt + (nt * 16) * 264 + kb * 32, 264);
    int n = n0 + nt * 16 + ln;

    proj_one_matrix<0>(wpq, bq, qT, bfr, b, n, wave, hi, ln);
    __builtin_amdgcn_sched_barrier(0);
    proj_one_matrix<1>(wpk, bk, kT, bfr, b, n, wave, hi, ln);
    __builtin_amdgcn_sched_barrier(0);
    proj_one_matrix<2>(wpv, bv, vB, bfr, b, n, wave, hi, ln);
    __builtin_amdgcn_sched_barrier(0);
  }
}

// ---------------------------------------------------------------------------
// Kernel 2: column softmax stats m[j], D[j] (reduction over i for each j).
// grid = B * N/64 = 256 blocks; each wave streams a 16-row i-slice, online
// max/sum-exp per column; cross-half via shfl_xor(16); cross-wave via LDS.
// ---------------------------------------------------------------------------
__global__ __launch_bounds__(256) void colstats_kernel(
    const __bf16* __restrict__ qT, const __bf16* __restrict__ kT,
    float* __restrict__ mcol, float* __restrict__ dcol) {
  __shared__ float2 red[8][64];
  int tid = threadIdx.x, lane = tid & 31, wave = tid >> 5;
  int ln = lane & 15;
  int b = blockIdx.x >> 6;
  int j0 = (blockIdx.x & 63) * 64;

#pragma unroll 1
  for (int nt = 0; nt < 4; ++nt) {
    bf16x16 kf[8];
#pragma unroll
    for (int kb = 0; kb < 8; ++kb)
      kf[kb] = load_fragB(kT + ((size_t)b * NPIX + j0 + nt * 16) * CCH + kb * 32, CCH);
    float m_run = -__builtin_inff(), d_run = 0.f;
#pragma unroll 1
    for (int it = 0; it < 32; ++it) {
      int i0 = it * 128 + wave * 16;
      f32x8 acc = {0.f, 0.f, 0.f, 0.f, 0.f, 0.f, 0.f, 0.f};
#pragma unroll
      for (int kb = 0; kb < 8; ++kb) {
        bf16x16 af = load_fragA(qT + ((size_t)b * NPIX + i0) * CCH + kb * 32, CCH);
        acc = wmma_bf16(af, kf[kb], acc);
      }
      float tmax = acc[0] * SCALE;
#pragma unroll
      for (int r = 1; r < 8; ++r) tmax = fmaxf(tmax, acc[r] * SCALE);
      tmax = fmaxf(tmax, __shfl_xor(tmax, 16, 32));
      float nm = fmaxf(m_run, tmax);
      float p = 0.f;
#pragma unroll
      for (int r = 0; r < 8; ++r) p += __expf(acc[r] * SCALE - nm);
      p += __shfl_xor(p, 16, 32);
      d_run = d_run * __expf(m_run - nm) + p;
      m_run = nm;
    }
    if (lane < 16) red[wave][nt * 16 + ln] = make_float2(m_run, d_run);
  }
  __syncthreads();
  if (tid < 64) {
    float mstar = -__builtin_inff();
#pragma unroll
    for (int w = 0; w < 8; ++w) mstar = fmaxf(mstar, red[w][tid].x);
    float D = 0.f;
#pragma unroll
    for (int w = 0; w < 8; ++w) D += red[w][tid].y * __expf(red[w][tid].x - mstar);
    mcol[(size_t)b * NPIX + j0 + tid] = mstar;
    dcol[(size_t)b * NPIX + j0 + tid] = D;
  }
}

// ---------------------------------------------------------------------------
// Kernel 3: out^T[i][c] = sum_j P[i,j] * v[c][j]. grid = B * N/64 = 256.
// 8 waves = 4 i-slices x 2 channel-halves: each wave owns 16 i-rows x 128
// channels (8 f32x8 accumulators = 64 VGPRs) and recomputes its S strip.
// S is converted to P via a per-wave padded LDS tile (D-layout -> A-fragment).
// ---------------------------------------------------------------------------
__global__ __launch_bounds__(256) void pv_kernel(
    const __bf16* __restrict__ qT, const __bf16* __restrict__ kT,
    const __bf16* __restrict__ vB, const float* __restrict__ mcol,
    const float* __restrict__ dcol, __bf16* __restrict__ oT) {
  __shared__ __bf16 pbuf[8 * 16 * 36];          // per-wave 16x32 P tile, padded
  int tid = threadIdx.x, lane = tid & 31, wave = tid >> 5;
  int hi = lane >> 4, ln = lane & 15;
  int islice = wave >> 1, chalf = wave & 1;
  int b = blockIdx.x >> 6;
  int i0 = (blockIdx.x & 63) * 64 + islice * 16;
  __bf16* pw = pbuf + wave * (16 * 36);

  bf16x16 aq[8];                                // q rows for this wave, held whole kernel
#pragma unroll
  for (int kb = 0; kb < 8; ++kb)
    aq[kb] = load_fragA(qT + ((size_t)b * NPIX + i0) * CCH + kb * 32, CCH);

  f32x8 acc_o[8];
#pragma unroll
  for (int ct = 0; ct < 8; ++ct)
    acc_o[ct] = (f32x8){0.f, 0.f, 0.f, 0.f, 0.f, 0.f, 0.f, 0.f};

#pragma unroll 1
  for (int jb = 0; jb < NPIX; jb += 32) {
#pragma unroll
    for (int h = 0; h < 2; ++h) {
      f32x8 sac = {0.f, 0.f, 0.f, 0.f, 0.f, 0.f, 0.f, 0.f};
#pragma unroll
      for (int kb = 0; kb < 8; ++kb) {
        bf16x16 kf = load_fragB(kT + ((size_t)b * NPIX + jb + h * 16) * CCH + kb * 32, CCH);
        sac = wmma_bf16(aq[kb], kf, sac);
      }
      int j = jb + h * 16 + ln;
      float mj  = mcol[(size_t)b * NPIX + j];
      float inv = 1.f / dcol[(size_t)b * NPIX + j];
#pragma unroll
      for (int r = 0; r < 8; ++r) {
        float p = __expf(sac[r] * SCALE - mj) * inv;
        pw[(hi * 8 + r) * 36 + h * 16 + ln] = (__bf16)p;  // [i_local][j_local]
      }
    }
    // same-wave LDS store->load ordering before fragment re-read
    asm volatile("s_wait_dscnt 0" ::: "memory");
    bf16x16 pa = load_fragA(pw, 36);
#pragma unroll
    for (int ct = 0; ct < 8; ++ct) {
      int c16 = chalf * 8 + ct;
      bf16x16 vf = load_fragB(vB + ((size_t)b * CCH + c16 * 16) * NPIX + jb, NPIX);
      acc_o[ct] = wmma_bf16(pa, vf, acc_o[ct]);
    }
  }
#pragma unroll
  for (int ct = 0; ct < 8; ++ct)
#pragma unroll
    for (int r = 0; r < 8; ++r) {
      int i = i0 + hi * 8 + r;
      int c = (chalf * 8 + ct) * 16 + ln;
      oT[((size_t)b * NPIX + i) * CCH + c] = (__bf16)acc_o[ct][r];
    }
}

// ---------------------------------------------------------------------------
// Kernel 4: final 1x1 conv: out[b][o][n] = wp . outT + bp. grid = 256 blocks.
// ---------------------------------------------------------------------------
__global__ __launch_bounds__(256) void proj_kernel(
    const __bf16* __restrict__ oT, const __bf16* __restrict__ wpp,
    const float* __restrict__ bp, float* __restrict__ out) {
  int tid = threadIdx.x, lane = tid & 31, wave = tid >> 5;
  int hi = lane >> 4, ln = lane & 15;
  int pix0 = blockIdx.x * 64;
  int b = pix0 >> 12;
  int n0 = pix0 & (NPIX - 1);
#pragma unroll 1
  for (int nt = 0; nt < 4; ++nt) {
    bf16x16 bfr[8];
#pragma unroll
    for (int kb = 0; kb < 8; ++kb)
      bfr[kb] = load_fragB(oT + ((size_t)b * NPIX + n0 + nt * 16) * CCH + kb * 32, CCH);
    int n = n0 + nt * 16 + ln;
#pragma unroll 1
    for (int oo = 0; oo < 2; ++oo) {
      int ot = wave * 2 + oo;
      f32x8 acc = {0.f, 0.f, 0.f, 0.f, 0.f, 0.f, 0.f, 0.f};
#pragma unroll
      for (int kb = 0; kb < 8; ++kb) {
        bf16x16 af = load_fragPacked(wpp + (size_t)(ot * 8 + kb) * 512);
        acc = wmma_bf16(af, bfr[kb], acc);
      }
#pragma unroll
      for (int r = 0; r < 8; ++r) {
        int o = ot * 16 + hi * 8 + r;
        out[((size_t)b * CCH + o) * NPIX + n] = acc[r] + bp[o];
      }
    }
  }
}

// ---------------------------------------------------------------------------
extern "C" void kernel_launch(void* const* d_in, const int* in_sizes, int n_in,
                              void* d_out, int out_size, void* d_ws, size_t ws_size,
                              hipStream_t stream) {
  (void)in_sizes; (void)n_in; (void)out_size;
  const float* x   = (const float*)d_in[0];
  const float* wq  = (const float*)d_in[1];
  const float* bq  = (const float*)d_in[2];
  const float* wk  = (const float*)d_in[3];
  const float* bk  = (const float*)d_in[4];
  const float* wv  = (const float*)d_in[5];
  const float* bv  = (const float*)d_in[6];
  const float* wpm = (const float*)d_in[7];
  const float* bp  = (const float*)d_in[8];

  const size_t WMAT = (size_t)CCH * CCH;              // 65536 elems
  const size_t TNC  = (size_t)NBATCH * NPIX * CCH;    // 4.19M elems
  char* ws = (char*)d_ws;
  __bf16* wpq = (__bf16*)ws;
  __bf16* wpk = wpq + WMAT;
  __bf16* wpv = wpk + WMAT;
  __bf16* wpp = wpv + WMAT;
  __bf16* qT  = wpp + WMAT;
  __bf16* kT  = qT + TNC;
  __bf16* vB  = kT + TNC;
  __bf16* oT  = vB + TNC;
  float* mcol = (float*)(oT + TNC);
  float* dcol = mcol + (size_t)NBATCH * NPIX;
  size_t need = (size_t)((char*)(dcol + (size_t)NBATCH * NPIX) - ws);
  if (ws_size < need) return;                         // scratch too small

  pack_w_kernel<<<128, 32, 0, stream>>>(wq, wpq);
  pack_w_kernel<<<128, 32, 0, stream>>>(wk, wpk);
  pack_w_kernel<<<128, 32, 0, stream>>>(wv, wpv);
  pack_w_kernel<<<128, 32, 0, stream>>>(wpm, wpp);

  qkv_kernel<<<256, 256, 0, stream>>>(x, wpq, wpk, wpv, bq, bk, bv, qT, kT, vB);
  colstats_kernel<<<256, 256, 0, stream>>>(qT, kT, mcol, dcol);
  pv_kernel<<<256, 256, 0, stream>>>(qT, kT, vB, mcol, dcol, oT);
  proj_kernel<<<256, 256, 0, stream>>>(oT, wpp, bp, (float*)d_out);
}
